// ODEFunc_32315333935910
// MI455X (gfx1250) — matrix operand order repaired and checked by reading, not compiled, not run on hardware
//
#include <hip/hip_runtime.h>
#include <math.h>

#define BDIM 3
#define CONDD 64
#define HID 128
#define INPD 68
#define KIN 96           // K of layer-1 padded to 3 x 32
#define MT 64            // batch rows per workgroup
#define NTHREADS 256     // 8 wave32

typedef __attribute__((ext_vector_type(16))) _Float16 v16h;
typedef __attribute__((ext_vector_type(8)))  _Float16 v8h;
typedef __attribute__((ext_vector_type(8)))  float    v8f;

struct SMem {
  _Float16 Winc[HID][KIN];   // Winc[n][k] = W_in[k][n] (k>=68 zero) : fwd B operand
  _Float16 Wh1c[HID][HID];   // Wh1c[n][k] = W_h1[k][n]              : fwd B operand
  _Float16 Wh2c[HID][HID];
  _Float16 Wh1n[HID][HID];   // natural row-major W_h1                : bwd B operand (W^T)
  _Float16 Wh2n[HID][HID];
  float    Woutf[HID][BDIM]; // f32: tiny out layer + eps@Wout^T done in VALU
  float    Winzf[BDIM][HID]; // first 3 rows of W_in (z-slice of W_in^T)
  float    bin[HID];
  float    bh1[HID];
  float    bh2[HID];
  float    bout[4];
  _Float16 inp[MT][KIN];     // [z | context | t | 0-pad]
  _Float16 h1[MT][HID];
  _Float16 h2[MT][HID];
  _Float16 h3[MT][HID];
  _Float16 g [MT][HID];      // gradient tile, reused
  float    epsF[MT][4];
};

// branch-free tanh: v_mul + v_exp + v_add + v_rcp + v_fma; clamp avoids overflow.
__device__ inline float tanh_fast(float x) {
  float xc = fminf(fmaxf(x, -15.f), 15.f);
  float e  = __expf(2.f * xc);                 // v_exp_f32 path
  float r  = __builtin_amdgcn_rcpf(e + 1.f);   // v_rcp_f32
  return 1.f - 2.f * r;
}

// A fragment 16x32 f16 from row-major src (ld elems per row), tile at (m0,k0).
// Per-lane data is two contiguous 16B runs -> two ds_load_b128.
__device__ inline v16h load_a(const _Float16* src, int ld, int m0, int k0, int lane) {
  int m = m0 + (lane & 15);
  int kb = k0 + ((lane & 16) ? 8 : 0);
  const v8h* p = (const v8h*)(src + m * ld + kb);   // 16B aligned
  v8h lo = p[0];   // K = kb .. kb+7   -> VGPR 0..3
  v8h hi = p[2];   // K = kb+16..kb+23 -> VGPR 4..7
  return __builtin_shufflevector(lo, hi, 0, 1, 2, 3, 4, 5, 6, 7,
                                 8, 9, 10, 11, 12, 13, 14, 15);
}

// B fragment 32x16: element (k,n) read from src[n*ld + k] (k contiguous).
// Per-lane data is two contiguous 16B runs -> two ds_load_b128.
__device__ inline v16h load_b(const _Float16* src, int ld, int k0, int n0, int lane) {
  int n = n0 + (lane & 15);
  int kb = k0 + ((lane & 16) ? 8 : 0);
  const v8h* p = (const v8h*)(src + n * ld + kb);   // 16B aligned
  v8h lo = p[0];   // K = kb..kb+7
  v8h hi = p[2];   // K = kb+16..kb+23
  return __builtin_shufflevector(lo, hi, 0, 1, 2, 3, 4, 5, 6, 7,
                                 8, 9, 10, 11, 12, 13, 14, 15);
}

__device__ inline void zero_acc(v8f acc[4]) {
#pragma unroll
  for (int i = 0; i < 4; ++i) {
    v8f z = {0.f, 0.f, 0.f, 0.f, 0.f, 0.f, 0.f, 0.f};
    acc[i] = z;
  }
}

// wave computes 4 adjacent 16x16 C tiles: rows mt*16.., cols (nb..nb+3)*16
template <int KSTEPS>
__device__ inline void gemm_tiles(const _Float16* A, int lda, const _Float16* Bm, int ldb,
                                  int mt, int nb, int lane, v8f acc[4]) {
#pragma unroll
  for (int ks = 0; ks < KSTEPS; ++ks) {
    v16h a = load_a(A, lda, mt * 16, ks * 32, lane);
#pragma unroll
    for (int i = 0; i < 4; ++i) {
      v16h b = load_b(Bm, ldb, ks * 32, (nb + i) * 16, lane);
      acc[i] = __builtin_amdgcn_wmma_f32_16x16x32_f16(
          false, a, false, b, (short)0, acc[i], false, false);
    }
  }
}

__device__ inline void store_tanh(_Float16* H, int ld, const float* bias,
                                  int mt, int nb, int lane, const v8f acc[4]) {
  int rbase = mt * 16 + ((lane & 16) ? 8 : 0);
  int cl = lane & 15;
#pragma unroll
  for (int i = 0; i < 4; ++i) {
    int col = (nb + i) * 16 + cl;
    float bb = bias[col];
#pragma unroll
    for (int r = 0; r < 8; ++r)
      H[(rbase + r) * ld + col] = (_Float16)tanh_fast(acc[i][r] + bb);
  }
}

__device__ inline void store_grad(_Float16* G, const _Float16* H, int ld,
                                  int mt, int nb, int lane, const v8f acc[4]) {
  int rbase = mt * 16 + ((lane & 16) ? 8 : 0);
  int cl = lane & 15;
#pragma unroll
  for (int i = 0; i < 4; ++i) {
    int col = (nb + i) * 16 + cl;
#pragma unroll
    for (int r = 0; r < 8; ++r) {
      int rr = rbase + r;
      float h = (float)H[rr * ld + col];
      G[rr * ld + col] = (_Float16)(acc[i][r] * (1.f - h * h));
    }
  }
}

__global__ __launch_bounds__(NTHREADS)
void odefunc_fused_kernel(const float* __restrict__ t, const float* __restrict__ z,
                          const float* __restrict__ context, const float* __restrict__ eps,
                          const float* __restrict__ W_in, const float* __restrict__ b_in,
                          const float* __restrict__ W_h1, const float* __restrict__ b_h1,
                          const float* __restrict__ W_h2, const float* __restrict__ b_h2,
                          const float* __restrict__ W_out, const float* __restrict__ b_out,
                          const float* __restrict__ out_scale, float* __restrict__ out, int Bn) {
  __shared__ SMem s;
  const int tid = threadIdx.x;
  const int wave = tid >> 5, lane = tid & 31;
  const int m0 = blockIdx.x * MT;
  const float tval = t[0];
  const float oscale = out_scale[0];

  // ---- stage weights into LDS (f16, both layouts) ----
  for (int i = tid; i < HID * KIN; i += NTHREADS) {
    int n = i / KIN, k = i % KIN;
    s.Winc[n][k] = (k < INPD) ? (_Float16)W_in[k * HID + n] : (_Float16)0.f;
  }
  for (int i = tid; i < HID * HID; i += NTHREADS) {
    int r = i / HID, c = i % HID;
    float w1 = W_h1[i], w2 = W_h2[i];
    s.Wh1n[r][c] = (_Float16)w1;  s.Wh2n[r][c] = (_Float16)w2;
    s.Wh1c[c][r] = (_Float16)w1;  s.Wh2c[c][r] = (_Float16)w2;
  }
  for (int i = tid; i < HID * BDIM; i += NTHREADS) s.Woutf[i / BDIM][i % BDIM] = W_out[i];
  for (int i = tid; i < BDIM * HID; i += NTHREADS) s.Winzf[i / HID][i % HID] = W_in[i];
  for (int i = tid; i < HID; i += NTHREADS) { s.bin[i] = b_in[i]; s.bh1[i] = b_h1[i]; s.bh2[i] = b_h2[i]; }
  if (tid < BDIM) s.bout[tid] = b_out[tid];

  // ---- stage batch tile: inp = [z | context | t | 0] ----
  for (int i = tid; i < MT * KIN; i += NTHREADS) {
    int r = i / KIN, k = i % KIN;
    int row = m0 + r;
    float v = 0.f;
    if (row < Bn) {
      if (k < BDIM)                 v = z[(size_t)row * BDIM + k];
      else if (k < BDIM + CONDD)    v = context[(size_t)row * CONDD + (k - BDIM)];
      else if (k == BDIM + CONDD)   v = tval;
    }
    s.inp[r][k] = (_Float16)v;
  }
  for (int i = tid; i < MT * BDIM; i += NTHREADS) {
    int r = i / BDIM, d = i % BDIM;
    int row = m0 + r;
    s.epsF[r][d] = (row < Bn) ? eps[(size_t)row * BDIM + d] : 0.f;
  }
  __syncthreads();

  const int mt = wave >> 1;        // 0..3 : M tile
  const int nb = (wave & 1) * 4;   // n-tile base 0 or 4
  v8f acc[4];

  // ---- forward: 3 WMMA layers ----
  zero_acc(acc);
  gemm_tiles<KIN / 32>(&s.inp[0][0], KIN, &s.Winc[0][0], KIN, mt, nb, lane, acc);
  store_tanh(&s.h1[0][0], HID, s.bin, mt, nb, lane, acc);
  __syncthreads();

  zero_acc(acc);
  gemm_tiles<HID / 32>(&s.h1[0][0], HID, &s.Wh1c[0][0], HID, mt, nb, lane, acc);
  store_tanh(&s.h2[0][0], HID, s.bh1, mt, nb, lane, acc);
  __syncthreads();

  zero_acc(acc);
  gemm_tiles<HID / 32>(&s.h2[0][0], HID, &s.Wh2c[0][0], HID, mt, nb, lane, acc);
  store_tanh(&s.h3[0][0], HID, s.bh2, mt, nb, lane, acc);
  __syncthreads();

  // ---- out layer f (N=3, f32 VALU) ----
  for (int i = tid; i < MT * BDIM; i += NTHREADS) {
    int r = i / BDIM, d = i % BDIM;
    int row = m0 + r;
    float a = s.bout[d];
    for (int k = 0; k < HID; ++k) a += (float)s.h3[r][k] * s.Woutf[k][d];
    if (row < Bn) out[(size_t)row * BDIM + d] = a * oscale;
  }
  // ---- g3 = (eps*scale) @ W_out^T, times tanh'(h3) ----
  for (int i = tid; i < MT * HID; i += NTHREADS) {
    int r = i / HID, n = i % HID;
    float gsum = 0.f;
    for (int d = 0; d < BDIM; ++d) gsum += s.epsF[r][d] * s.Woutf[n][d];
    gsum *= oscale;
    float h = (float)s.h3[r][n];
    s.g[r][n] = (_Float16)(gsum * (1.f - h * h));
  }
  __syncthreads();

  // ---- backward WMMA: g2 = (g3 @ W_h2^T) * tanh'(h2) ----
  zero_acc(acc);
  gemm_tiles<HID / 32>(&s.g[0][0], HID, &s.Wh2n[0][0], HID, mt, nb, lane, acc);
  __syncthreads();                           // all reads of g done
  store_grad(&s.g[0][0], &s.h2[0][0], HID, mt, nb, lane, acc);
  __syncthreads();

  // ---- backward WMMA: g1 = (g2 @ W_h1^T) * tanh'(h1) ----
  zero_acc(acc);
  gemm_tiles<HID / 32>(&s.g[0][0], HID, &s.Wh1n[0][0], HID, mt, nb, lane, acc);
  __syncthreads();
  store_grad(&s.g[0][0], &s.h1[0][0], HID, mt, nb, lane, acc);
  __syncthreads();

  // ---- gz = g1 @ (W_in^T)[:, 0:3]; neg_div = -(gz . eps) ----
  if (tid < MT) {
    int r = tid;
    int row = m0 + r;
    float nd = 0.f;
    for (int d = 0; d < BDIM; ++d) {
      float a = 0.f;
      for (int k = 0; k < HID; ++k) a += (float)s.g[r][k] * s.Winzf[d][k];
      nd += a * s.epsF[r][d];
    }
    if (row < Bn) out[(size_t)Bn * BDIM + row] = -nd;
  }
}

extern "C" void kernel_launch(void* const* d_in, const int* in_sizes, int n_in,
                              void* d_out, int out_size, void* d_ws, size_t ws_size,
                              hipStream_t stream) {
  const float* t       = (const float*)d_in[0];
  const float* z       = (const float*)d_in[1];
  // d_in[2] = logp (unused by the reference computation)
  const float* context = (const float*)d_in[3];
  const float* eps     = (const float*)d_in[4];
  const float* W_in    = (const float*)d_in[5];
  const float* b_in    = (const float*)d_in[6];
  const float* W_h1    = (const float*)d_in[7];
  const float* b_h1    = (const float*)d_in[8];
  const float* W_h2    = (const float*)d_in[9];
  const float* b_h2    = (const float*)d_in[10];
  const float* W_out   = (const float*)d_in[11];
  const float* b_out   = (const float*)d_in[12];
  const float* oscale  = (const float*)d_in[13];
  int Bn = in_sizes[1] / BDIM;
  int grid = (Bn + MT - 1) / MT;
  hipLaunchKernelGGL(odefunc_fused_kernel, dim3(grid), dim3(NTHREADS), 0, stream,
                     t, z, context, eps, W_in, b_in, W_h1, b_h1, W_h2, b_h2,
                     W_out, b_out, oscale, (float*)d_out, Bn);
}
